// TransformerBlock_83691732730567
// MI455X (gfx1250) — matrix operand order repaired and checked by reading, not compiled, not run on hardware
//
#include <hip/hip_runtime.h>

typedef _Float16 f16;
typedef __attribute__((ext_vector_type(16))) _Float16 v16h;
typedef __attribute__((ext_vector_type(8)))  _Float16 v8h;
typedef __attribute__((ext_vector_type(4)))  _Float16 v4h;
typedef __attribute__((ext_vector_type(8)))  float    v8f;

#define B_  2
#define N_  2048
#define D_  1024
#define H_  16
#define DH_ 64
#define M_  (B_*N_)   // 4096 token rows

// ---------------- f32 -> f16 conversion (vectorized) ----------------
__global__ __launch_bounds__(256)
void cvt_f16_kernel(const float* __restrict__ in, f16* __restrict__ out, int n4) {
    int i = blockIdx.x * blockDim.x + threadIdx.x;
    if (i < n4) {
        const float4 v = ((const float4*)in)[i];
        v4h h;
        h[0] = (f16)v.x; h[1] = (f16)v.y; h[2] = (f16)v.z; h[3] = (f16)v.w;
        ((v4h*)out)[i] = h;
    }
}

// ---------------- fused QKV projection GEMM ----------------
// out[t, m] = sum_k xh[t,k] * W[m,k] + bias[m]   (W row-major (D,D), K contiguous both sides)
// z = blockIdx.z selects Q / K / V.  Q,K stored row-major f16 (B,N,D); V stored transposed (B,D,N).
__global__ __launch_bounds__(128)
void qkv_gemm_kernel(const f16* __restrict__ xh,
                     const f16* __restrict__ wq, const f16* __restrict__ wk, const f16* __restrict__ wv,
                     const float* __restrict__ bq, const float* __restrict__ bk, const float* __restrict__ bv,
                     f16* __restrict__ Qh, f16* __restrict__ Kh, f16* __restrict__ Vth)
{
    const int z = blockIdx.z;
    const f16*   W    = (z == 0) ? wq : (z == 1) ? wk : wv;
    const float* bias = (z == 0) ? bq : (z == 1) ? bk : bv;

    const int lane = threadIdx.x & 31;
    const int wave = threadIdx.x >> 5;
    const int half = lane >> 4;   // 0: lanes 0-15, 1: lanes 16-31
    const int l16  = lane & 15;

    const int m0 = blockIdx.x * 128 + (wave & 1) * 64;   // token-row base of this wave's tile
    const int n0 = blockIdx.y * 128 + (wave >> 1) * 64;  // out-col base

    const int kbA = half ? 8 : 0;    // A-fragment K sub-base (16-bit A layout)
    const int kbB = half ? 8 : 0;    // B-fragment K sub-base

    v8f acc[4][4];
    {
        v8f zero = {};
        #pragma unroll
        for (int i = 0; i < 4; i++)
            #pragma unroll
            for (int j = 0; j < 4; j++) acc[i][j] = zero;
    }

    for (int k0 = 0; k0 < D_; k0 += 32) {
        v16h a[4], bf[4];
        #pragma unroll
        for (int i = 0; i < 4; i++) {
            const f16* p = xh + (size_t)(m0 + i * 16 + l16) * D_ + k0 + kbA;
            v8h lo = *(const v8h*)p;
            v8h hi = *(const v8h*)(p + 16);
            #pragma unroll
            for (int t = 0; t < 8; t++) { a[i][t] = lo[t]; a[i][8 + t] = hi[t]; }
        }
        #pragma unroll
        for (int j = 0; j < 4; j++) {
            const f16* p = W + (size_t)(n0 + j * 16 + l16) * D_ + k0 + kbB;
            v8h lo = *(const v8h*)p;
            v8h hi = *(const v8h*)(p + 16);
            #pragma unroll
            for (int t = 0; t < 8; t++) { bf[j][t] = lo[t]; bf[j][8 + t] = hi[t]; }
        }
        #pragma unroll
        for (int i = 0; i < 4; i++)
            #pragma unroll
            for (int j = 0; j < 4; j++)
                acc[i][j] = __builtin_amdgcn_wmma_f32_16x16x32_f16(
                    false, a[i], false, bf[j], (short)0, acc[i][j], false, false);
    }

    if (z < 2) {
        f16* out = (z == 0) ? Qh : Kh;
        #pragma unroll
        for (int j = 0; j < 4; j++) {
            const float bcol = bias[n0 + j * 16 + l16];
            #pragma unroll
            for (int i = 0; i < 4; i++) {
                const int row = m0 + i * 16 + half * 8;
                f16* op = out + (size_t)row * D_ + n0 + j * 16 + l16;
                #pragma unroll
                for (int r = 0; r < 8; r++)
                    op[(size_t)r * D_] = (f16)(acc[i][j][r] + bcol);
            }
        }
    } else {
        // Vt (B, D, N): C-fragment rows (consecutive M per VGPR) pack into one b128 store.
        #pragma unroll
        for (int j = 0; j < 4; j++) {
            const int d = n0 + j * 16 + l16;
            const float bcol = bias[d];
            #pragma unroll
            for (int i = 0; i < 4; i++) {
                const int t0 = m0 + i * 16 + half * 8;   // 8 consecutive tokens
                const int b  = t0 >> 11;
                const int n  = t0 & (N_ - 1);
                v8h pk;
                #pragma unroll
                for (int r = 0; r < 8; r++) pk[r] = (f16)(acc[i][j][r] + bcol);
                *(v8h*)(Vth + ((size_t)b * D_ + d) * N_ + n) = pk;
            }
        }
    }
}

// ---------------- flash-style attention ----------------
// grid: (N/64, H, B), block: 128 (4 waves). Each wave owns a 16-row q tile.
__global__ __launch_bounds__(128)
void attn_kernel(const f16* __restrict__ Qh, const f16* __restrict__ Kh,
                 const f16* __restrict__ Vth, float* __restrict__ ctx)
{
    __shared__ __align__(16) f16 Pbuf[4][16][32];   // per-wave P staging (C-layout -> A-layout)

    const int lane = threadIdx.x & 31;
    const int wave = threadIdx.x >> 5;
    const int half = lane >> 4;
    const int l16  = lane & 15;

    const int b  = blockIdx.z;
    const int h  = blockIdx.y;
    const int q0 = blockIdx.x * 64 + wave * 16;

    const int kbA = half ? 8 : 0;
    const int kbB = half ? 8 : 0;

    // Q A-fragments over d (pre-scaled by 1/sqrt(DH) = 1/8, exact power of two)
    v16h qa[2];
    {
        const f16* qrow = Qh + (size_t)(b * N_ + q0 + l16) * D_ + h * DH_;
        #pragma unroll
        for (int s = 0; s < 2; s++) {
            const f16* p = qrow + s * 32 + kbA;
            v8h lo = *(const v8h*)p;
            v8h hi = *(const v8h*)(p + 16);
            #pragma unroll
            for (int t = 0; t < 8; t++) {
                qa[s][t]     = lo[t] * (f16)0.125f;
                qa[s][8 + t] = hi[t] * (f16)0.125f;
            }
        }
    }

    float mrow[8], lrow[8];
    v8f cacc[4];
    {
        v8f zero = {};
        #pragma unroll
        for (int r = 0; r < 8; r++) { mrow[r] = -3.0e38f; lrow[r] = 0.f; }
        #pragma unroll
        for (int dt = 0; dt < 4; dt++) cacc[dt] = zero;
    }

    for (int kx0 = 0; kx0 < N_; kx0 += 32) {
        // --- scores: two 16x16 tiles (contraction over d = 64 -> 2 WMMAs each) ---
        v8f s[2];
        #pragma unroll
        for (int t2 = 0; t2 < 2; t2++) {
            v8f sa = {};
            const f16* krow = Kh + (size_t)(b * N_ + kx0 + t2 * 16 + l16) * D_ + h * DH_;
            #pragma unroll
            for (int st = 0; st < 2; st++) {
                const f16* p = krow + st * 32 + kbB;
                v8h lo = *(const v8h*)p;
                v8h hi = *(const v8h*)(p + 16);
                v16h bf;
                #pragma unroll
                for (int t = 0; t < 8; t++) { bf[t] = lo[t]; bf[8 + t] = hi[t]; }
                sa = __builtin_amdgcn_wmma_f32_16x16x32_f16(
                    false, qa[st], false, bf, (short)0, sa, false, false);
            }
            s[t2] = sa;
        }

        // --- mask: replace with -1e9 where j <= i (attend only strictly-future) ---
        #pragma unroll
        for (int t2 = 0; t2 < 2; t2++) {
            const int j = kx0 + t2 * 16 + l16;
            #pragma unroll
            for (int r = 0; r < 8; r++) {
                const int i = q0 + r + half * 8;
                if (j <= i) s[t2][r] = -1.0e9f;
            }
        }

        // --- online softmax: row max (butterfly over the 16-lane group) ---
        float nm[8], sc[8];
        #pragma unroll
        for (int r = 0; r < 8; r++) {
            float v = fmaxf(s[0][r], s[1][r]);
            #pragma unroll
            for (int m = 1; m < 16; m <<= 1)
                v = fmaxf(v, __shfl_xor(v, m, 16));
            nm[r]   = fmaxf(mrow[r], v);
            sc[r]   = __expf(mrow[r] - nm[r]);
            mrow[r] = nm[r];
        }

        // P = exp(s - m)
        #pragma unroll
        for (int t2 = 0; t2 < 2; t2++)
            #pragma unroll
            for (int r = 0; r < 8; r++)
                s[t2][r] = __expf(s[t2][r] - nm[r]);

        // row sums + state update
        #pragma unroll
        for (int r = 0; r < 8; r++) {
            float v = s[0][r] + s[1][r];
            #pragma unroll
            for (int m = 1; m < 16; m <<= 1)
                v += __shfl_xor(v, m, 16);
            lrow[r] = lrow[r] * sc[r] + v;
        }
        #pragma unroll
        for (int dt = 0; dt < 4; dt++)
            #pragma unroll
            for (int r = 0; r < 8; r++)
                cacc[dt][r] *= sc[r];

        // --- P: C-layout -> LDS row-major [16][32] (same-wave LDS is in-order) ---
        #pragma unroll
        for (int t2 = 0; t2 < 2; t2++)
            #pragma unroll
            for (int r = 0; r < 8; r++)
                Pbuf[wave][r + half * 8][t2 * 16 + l16] = (f16)s[t2][r];
        asm volatile("" ::: "memory");

        // reload P as a WMMA A-fragment (16 x 32 over kx)
        v16h pa;
        {
            const f16* pp = &Pbuf[wave][l16][kbA];
            v8h lo = *(const v8h*)pp;
            v8h hi = *(const v8h*)(pp + 16);
            #pragma unroll
            for (int t = 0; t < 8; t++) { pa[t] = lo[t]; pa[8 + t] = hi[t]; }
        }

        // --- ctx += P @ V : B-fragments contiguous thanks to transposed V ---
        #pragma unroll
        for (int dt = 0; dt < 4; dt++) {
            const f16* vp = Vth + ((size_t)b * D_ + h * DH_ + dt * 16 + l16) * N_ + kx0 + kbB;
            v8h lo = *(const v8h*)vp;
            v8h hi = *(const v8h*)(vp + 16);
            v16h bf;
            #pragma unroll
            for (int t = 0; t < 8; t++) { bf[t] = lo[t]; bf[8 + t] = hi[t]; }
            cacc[dt] = __builtin_amdgcn_wmma_f32_16x16x32_f16(
                false, pa, false, bf, (short)0, cacc[dt], false, false);
        }
    }

    // normalize and store ctx (f32, (B,N,D))
    #pragma unroll
    for (int r = 0; r < 8; r++) lrow[r] = 1.0f / lrow[r];
    #pragma unroll
    for (int dt = 0; dt < 4; dt++)
        #pragma unroll
        for (int r = 0; r < 8; r++) {
            const int row = q0 + r + half * 8;
            ctx[((size_t)(b * N_ + row)) * D_ + h * DH_ + dt * 16 + l16] = cacc[dt][r] * lrow[r];
        }
}

// ---------------- residual + LayerNorm ----------------
__global__ __launch_bounds__(256)
void ln_kernel(const float* __restrict__ ctx, const float* __restrict__ x,
               const float* __restrict__ g, const float* __restrict__ bta,
               float* __restrict__ out)
{
    const int row = blockIdx.x;  // 0 .. B*N-1
    const float* cr = ctx + (size_t)row * D_;
    const float* xr = x   + (size_t)row * D_;
    const int tid = threadIdx.x;

    float y[4];
    float sum = 0.f, ss = 0.f;
    #pragma unroll
    for (int t = 0; t < 4; t++) {
        const int c = tid + t * 256;
        const float v = cr[c] + xr[c];
        y[t] = v; sum += v; ss += v * v;
    }

    __shared__ float red[2][8];
    #pragma unroll
    for (int m = 1; m < 32; m <<= 1) {
        sum += __shfl_xor(sum, m, 32);
        ss  += __shfl_xor(ss,  m, 32);
    }
    const int wv = tid >> 5, ln = tid & 31;
    if (ln == 0) { red[0][wv] = sum; red[1][wv] = ss; }
    __syncthreads();
    if (wv == 0) {
        float a = (ln < 8) ? red[0][ln] : 0.f;
        float q = (ln < 8) ? red[1][ln] : 0.f;
        #pragma unroll
        for (int m = 1; m < 8; m <<= 1) {
            a += __shfl_xor(a, m, 32);
            q += __shfl_xor(q, m, 32);
        }
        if (ln == 0) { red[0][0] = a; red[1][0] = q; }
    }
    __syncthreads();
    const float mean = red[0][0] * (1.0f / D_);
    const float var  = red[1][0] * (1.0f / D_) - mean * mean;
    const float rstd = rsqrtf(var + 1e-5f);
    #pragma unroll
    for (int t = 0; t < 4; t++) {
        const int c = tid + t * 256;
        out[(size_t)row * D_ + c] = (y[t] - mean) * rstd * g[c] + bta[c];
    }
}

extern "C" void kernel_launch(void* const* d_in, const int* in_sizes, int n_in,
                              void* d_out, int out_size, void* d_ws, size_t ws_size,
                              hipStream_t stream)
{
    (void)in_sizes; (void)n_in; (void)out_size; (void)ws_size;

    const float* x   = (const float*)d_in[0];
    const float* Wq  = (const float*)d_in[1];
    const float* bq  = (const float*)d_in[2];
    const float* Wk  = (const float*)d_in[3];
    const float* bk  = (const float*)d_in[4];
    const float* Wv  = (const float*)d_in[5];
    const float* bv  = (const float*)d_in[6];
    const float* lng = (const float*)d_in[7];
    const float* lnb = (const float*)d_in[8];
    float* out = (float*)d_out;

    // workspace layout (needs ~54 MB)
    char* ws = (char*)d_ws;
    size_t off = 0;
    f16* xh  = (f16*)(ws + off); off += (size_t)M_ * D_ * 2;   // 8 MB
    f16* wh0 = (f16*)(ws + off); off += (size_t)D_ * D_ * 2;   // 2 MB
    f16* wh1 = (f16*)(ws + off); off += (size_t)D_ * D_ * 2;
    f16* wh2 = (f16*)(ws + off); off += (size_t)D_ * D_ * 2;
    f16* Qh  = (f16*)(ws + off); off += (size_t)M_ * D_ * 2;   // 8 MB
    f16* Kh  = (f16*)(ws + off); off += (size_t)M_ * D_ * 2;   // 8 MB
    f16* Vth = (f16*)(ws + off); off += (size_t)M_ * D_ * 2;   // 8 MB (transposed (B,D,N))
    float* ctx = (float*)(ws + off);                           // 16 MB

    // f32 -> f16 conversions
    {
        const int n4x = (M_ * D_) / 4;
        cvt_f16_kernel<<<(n4x + 255) / 256, 256, 0, stream>>>(x, xh, n4x);
        const int n4w = (D_ * D_) / 4;
        cvt_f16_kernel<<<(n4w + 255) / 256, 256, 0, stream>>>(Wq, wh0, n4w);
        cvt_f16_kernel<<<(n4w + 255) / 256, 256, 0, stream>>>(Wk, wh1, n4w);
        cvt_f16_kernel<<<(n4w + 255) / 256, 256, 0, stream>>>(Wv, wh2, n4w);
    }

    // fused Q/K/V projection (WMMA)
    qkv_gemm_kernel<<<dim3(M_ / 128, D_ / 128, 3), 128, 0, stream>>>(
        xh, wh0, wh1, wh2, bq, bk, bv, Qh, Kh, Vth);

    // attention (WMMA flash-style, online softmax)
    attn_kernel<<<dim3(N_ / 64, H_, B_), 128, 0, stream>>>(Qh, Kh, Vth, ctx);

    // residual + LayerNorm
    ln_kernel<<<M_, 256, 0, stream>>>(ctx, x, lng, lnb, out);
}
